// TimeIntervalAwareTransformer_71038759076170
// MI455X (gfx1250) — matrix operand order repaired and checked by reading, not compiled, not run on hardware
//
#include <hip/hip_runtime.h>
#include <hip/hip_bf16.h>

// ---------------------------------------------------------------------------
// Time-interval-aware transformer forward for MI455X (gfx1250, wave32, WMMA).
// All dense contractions run through v_wmma_f32_16x16x32_f16 (f16 in, f32 acc).
// Embedding einsums are factored through the 25/8-entry vocabularies so they
// become small GEMMs instead of [B,L,L,D] tensors.
// ---------------------------------------------------------------------------

typedef __attribute__((ext_vector_type(16))) _Float16 v16h;
typedef __attribute__((ext_vector_type(8)))  float    v8f;

namespace {
constexpr int BN   = 8;
constexpr int LL   = 160;
constexpr int DD   = 128;
constexpr int NPOI = 10000;
}

// pack 4 f32 into fragment elements [base..base+3] (base is compile-time)
__device__ __forceinline__ void cvt4(v16h& f, const int base, const float4 a)
{
    f[base + 0] = (_Float16)a.x;
    f[base + 1] = (_Float16)a.y;
    f[base + 2] = (_Float16)a.z;
    f[base + 3] = (_Float16)a.w;
}

// ---------------------------------------------------------------------------
// Generic WMMA GEMM: C[bz] = act( A[bz] * op(B[bz]) (+bias) (+Cin[bz]) )
// One wave computes a 16x32 output tile (two 16x16 WMMAs sharing the A frag).
// Requirements (guaranteed by the launch sites):
//   - K % 32 == 0
//   - all leading dimensions are multiples of 4 floats (16B-aligned rows)
// Out-of-range rows/cols are CLAMPED on load (branchless; only affects output
// elements that the masked epilogue never stores), so EXEC stays all-ones for
// the WMMAs and all loads are unconditional wide loads. TRANSB is a template
// parameter so the hot loop has no branches at all.
// ---------------------------------------------------------------------------
template <int TRANSB>
__global__ __launch_bounds__(32)
void k_gemm_wmma(const float* __restrict__ A, int lda, long long sA,
                 const float* __restrict__ Bm, int ldb, long long sB,
                 const float* __restrict__ bias,
                 const float* __restrict__ Cin,
                 float* __restrict__ C, int ldc, long long sC,
                 int M, int N, int K, int relu)
{
    const int lane = threadIdx.x;           // 0..31
    const int tm   = blockIdx.y * 16;
    const int tn   = blockIdx.x * 32;
    const long long bz = blockIdx.z;

    const float* Ab = A  + bz * sA;
    const float* Bb = Bm + bz * sB;

    // A fragment (16-bit A 16x32): lane m = lane&15, K-group = lane>>4.
    // Per lane, chunk elements are k = kgrp*16 + {0..15}.
    const int  mA   = lane & 15;
    const int  kgrp = lane >> 4;
    const int  rowA = min(tm + mA, M - 1);                // clamped load row
    const float* aptr = Ab + (long long)rowA * lda + kgrp * 16;

    // B fragment (16-bit B 32x16): n = lane&15, k = (lane>>4)*16 + e.
    const int  kofsB = kgrp * 16;
    const int  n0c = min(tn + (lane & 15), N - 1);        // clamped load cols
    const int  n1c = min(tn + 16 + (lane & 15), N - 1);

    v8f acc0 = {}, acc1 = {};

    for (int k0 = 0; k0 < K; k0 += 32) {
        // ---- A fragment: 4x b128 loads, f32->f16 packed converts
        v16h af;
        cvt4(af,  0, *(const float4*)(aptr + k0));
        cvt4(af,  4, *(const float4*)(aptr + k0 + 4));
        cvt4(af,  8, *(const float4*)(aptr + k0 + 8));
        cvt4(af, 12, *(const float4*)(aptr + k0 + 12));

        v16h bf0, bf1;
        if (TRANSB) {
            // contiguous 16 floats per lane -> 4x b128 each
            const float* bp0 = Bb + (long long)n0c * ldb + k0 + kofsB;
            const float* bp1 = Bb + (long long)n1c * ldb + k0 + kofsB;
            cvt4(bf0,  0, *(const float4*)(bp0));
            cvt4(bf0,  4, *(const float4*)(bp0 + 4));
            cvt4(bf0,  8, *(const float4*)(bp0 + 8));
            cvt4(bf0, 12, *(const float4*)(bp0 + 12));
            cvt4(bf1,  0, *(const float4*)(bp1));
            cvt4(bf1,  4, *(const float4*)(bp1 + 4));
            cvt4(bf1,  8, *(const float4*)(bp1 + 8));
            cvt4(bf1, 12, *(const float4*)(bp1 + 12));
        } else {
            // column loads, coalesced across lanes (lane n varies fastest)
            const float* bp = Bb + (long long)(k0 + kofsB) * ldb;
#pragma unroll
            for (int e = 0; e < 16; ++e) {
                bf0[e] = (_Float16)bp[(long long)e * ldb + n0c];
                bf1[e] = (_Float16)bp[(long long)e * ldb + n1c];
            }
        }
        acc0 = __builtin_amdgcn_wmma_f32_16x16x32_f16(false, af, false, bf0, (short)0, acc0, false, false);
        acc1 = __builtin_amdgcn_wmma_f32_16x16x32_f16(false, af, false, bf1, (short)0, acc1, false, false);
    }

    // C/D layout: element r of v8f -> row = r + 8*(lane>=16), col = lane&15
    const int col0  = tn + (lane & 15);
    const int col1  = col0 + 16;
    const int rbase = kgrp * 8;
    float* Cb = C + bz * sC;
    const float* Cinb = Cin ? (Cin + bz * sC) : nullptr;
    const float bias0 = (bias && col0 < N) ? bias[col0] : 0.f;
    const float bias1 = (bias && col1 < N) ? bias[col1] : 0.f;
#pragma unroll
    for (int r = 0; r < 8; ++r) {
        const int orow = tm + rbase + r;
        if (orow < M) {
            const long long rb = (long long)orow * ldc;
            if (col0 < N) {
                float v = acc0[r] + bias0;
                if (Cinb) v += Cinb[rb + col0];
                if (relu) v = fmaxf(v, 0.f);
                Cb[rb + col0] = v;
            }
            if (col1 < N) {
                float v = acc1[r] + bias1;
                if (Cinb) v += Cinb[rb + col1];
                if (relu) v = fmaxf(v, 0.f);
                Cb[rb + col1] = v;
            }
        }
    }
}

static inline void gemm(hipStream_t st,
                        const float* A, int lda, long long sA,
                        const float* Bm, int ldb, long long sB, int transB,
                        const float* bias, const float* Cin,
                        float* C, int ldc, long long sC,
                        int M, int N, int K, int batch, int relu)
{
    dim3 grid((N + 31) / 32, (M + 15) / 16, batch);
    if (transB)
        k_gemm_wmma<1><<<grid, dim3(32), 0, st>>>(A, lda, sA, Bm, ldb, sB,
                                                  bias, Cin, C, ldc, sC, M, N, K, relu);
    else
        k_gemm_wmma<0><<<grid, dim3(32), 0, st>>>(A, lda, sA, Bm, ldb, sB,
                                                  bias, Cin, C, ldc, sC, M, N, K, relu);
}

// ---------------------------------------------------------------------------
// Zero-padded embedding copy: out[32,128], rows >= T are zero. Lets the
// wsum @ emb GEMMs run with K=32 (no runtime-K guards in the hot loop).
// ---------------------------------------------------------------------------
__global__ __launch_bounds__(128)
void k_pad_emb(const float* __restrict__ emb, int T, float* __restrict__ out)
{
    const int r = blockIdx.x;          // 0..31
    const int d = threadIdx.x;         // 0..127
    out[r * DD + d] = (r < T) ? emb[r * DD + d] : 0.f;
}

// ---------------------------------------------------------------------------
// x[b,l,:] = src[b,l,:] + pe[b,:]   (faithful to the reference's batch-index
// positional-encoding quirk).
// ---------------------------------------------------------------------------
__global__ __launch_bounds__(128)
void k_add_pe(const float* __restrict__ src, float* __restrict__ x)
{
    const int row = blockIdx.x;        // b*L + l
    const int d   = threadIdx.x;       // 0..127
    const int b   = row / LL;
    const float p2  = (float)((d >> 1) * 2);
    const float div = expf(p2 * (-logf(10000.f) / (float)DD));
    const float ang = (float)b * div;
    const float pe  = (d & 1) ? cosf(ang) : sinf(ang);
    const long long idx = (long long)row * DD + d;
    x[idx] = src[idx] + pe;
}

// ---------------------------------------------------------------------------
// Attention softmax row: s = (qk + qh[h] + qw[w]) / sqrt(D), masked -> MASK_VAL,
// softmax over all j (matches reference: finite mask value, full-axis softmax).
// ---------------------------------------------------------------------------
__global__ __launch_bounds__(256)
void k_attn_softmax(const float* __restrict__ qk, const float* __restrict__ qh,
                    const float* __restrict__ qw, const int* __restrict__ hmat,
                    const int* __restrict__ wmat, const int* __restrict__ seqlens,
                    float* __restrict__ aw)
{
    const int row = blockIdx.x;                 // b*L + i
    const int b   = row / LL;
    const int i   = row - b * LL;
    const int j   = threadIdx.x;
    __shared__ float red[256];

    const float MASK_VAL = -4294967295.0f;      // -2^32 + 1
    const float scale    = 0.08838834764831845f; // 1/sqrt(128)

    const bool act = j < LL;
    float s = -3.4e38f;
    if (act) {
        const bool masked = (j > i) || (i >= seqlens[b]);
        if (masked) {
            s = MASK_VAL;
        } else {
            const long long idx = (long long)row * LL + j;
            s = (qk[idx] + qh[row * 32 + hmat[idx]] + qw[row * 32 + wmat[idx]]) * scale;
        }
    }
    red[threadIdx.x] = s; __syncthreads();
    for (int st = 128; st > 0; st >>= 1) {
        if (threadIdx.x < st) red[threadIdx.x] = fmaxf(red[threadIdx.x], red[threadIdx.x + st]);
        __syncthreads();
    }
    const float mx = red[0]; __syncthreads();
    const float e = act ? expf(s - mx) : 0.f;
    red[threadIdx.x] = e; __syncthreads();
    for (int st = 128; st > 0; st >>= 1) {
        if (threadIdx.x < st) red[threadIdx.x] += red[threadIdx.x + st];
        __syncthreads();
    }
    const float inv = 1.f / red[0];
    if (act) aw[(long long)row * LL + j] = e * inv;
}

// ---------------------------------------------------------------------------
// Decoder causal softmax: s = xu[b,j] + hu[b,lh] + wu[b,lw] for j<=i; exact
// zeros for j>i (reference uses -inf mask).
// ---------------------------------------------------------------------------
__global__ __launch_bounds__(256)
void k_dec_softmax(const float* __restrict__ xu, const float* __restrict__ hu,
                   const float* __restrict__ wu, const int* __restrict__ lh,
                   const int* __restrict__ lw, float* __restrict__ aw)
{
    const int row = blockIdx.x;
    const int b   = row / LL;
    const int i   = row - b * LL;
    const int j   = threadIdx.x;
    __shared__ float red[256];

    const bool act = (j < LL) && (j <= i);
    float s = -3.4e38f;
    if (act) {
        const long long idx = (long long)row * LL + j;
        s = xu[b * LL + j] + hu[b * 32 + lh[idx]] + wu[b * 32 + lw[idx]];
    }
    red[threadIdx.x] = s; __syncthreads();
    for (int st = 128; st > 0; st >>= 1) {
        if (threadIdx.x < st) red[threadIdx.x] = fmaxf(red[threadIdx.x], red[threadIdx.x + st]);
        __syncthreads();
    }
    const float mx = red[0]; __syncthreads();
    const float e = act ? expf(s - mx) : 0.f;
    red[threadIdx.x] = e; __syncthreads();
    for (int st = 128; st > 0; st >>= 1) {
        if (threadIdx.x < st) red[threadIdx.x] += red[threadIdx.x + st];
        __syncthreads();
    }
    const float inv = 1.f / red[0];
    if (j < LL) aw[(long long)row * LL + j] = act ? e * inv : 0.f;
}

// ---------------------------------------------------------------------------
// Bucketed weight sums: out[row,t] = sum_{j: mat[row,j]==t} aw[row,j], t<32.
// Lane t owns bucket t -> deterministic, no atomics. mat values < 25 (or 8).
// ---------------------------------------------------------------------------
__global__ __launch_bounds__(32)
void k_wsum(const float* __restrict__ aw, const int* __restrict__ mat,
            float* __restrict__ out)
{
    const int row = blockIdx.x;
    const int t   = threadIdx.x;                // bucket id, 0..31
    const long long base = (long long)row * LL;
    float acc = 0.f;
    for (int j = 0; j < LL; ++j)
        if (mat[base + j] == t) acc += aw[base + j];
    out[row * 32 + t] = acc;
}

// ---------------------------------------------------------------------------
// out = LayerNorm(xa + xr) * g + b     (eps = 1e-5, matches reference)
// ---------------------------------------------------------------------------
__global__ __launch_bounds__(128)
void k_add_ln(const float* __restrict__ xa, const float* __restrict__ xr,
              const float* __restrict__ g, const float* __restrict__ bb,
              float* __restrict__ out)
{
    const int row = blockIdx.x;
    const int t   = threadIdx.x;
    __shared__ float red[128];
    const long long idx = (long long)row * DD + t;
    const float v = xa[idx] + xr[idx];
    red[t] = v; __syncthreads();
    for (int st = 64; st > 0; st >>= 1) {
        if (t < st) red[t] += red[t + st];
        __syncthreads();
    }
    const float m = red[0] * (1.f / (float)DD); __syncthreads();
    const float d = v - m;
    red[t] = d * d; __syncthreads();
    for (int st = 64; st > 0; st >>= 1) {
        if (t < st) red[t] += red[t + st];
        __syncthreads();
    }
    const float var = red[0] * (1.f / (float)DD);
    out[idx] = d * rsqrtf(var + 1e-5f) * g[t] + bb[t];
}

// ---------------------------------------------------------------------------
// One attention block (all GEMMs via WMMA). hpad/wpad are zero-padded [32,128]
// copies of hour_emb/day_emb.
// ---------------------------------------------------------------------------
static void run_attn_block(hipStream_t st, const float* xin, float* xout,
    const float* Wq, const float* bq, const float* Wk, const float* bk,
    const float* Wv, const float* bv,
    const float* W1, const float* b1, const float* W2, const float* b2,
    const float* g1, const float* be1, const float* g2, const float* be2,
    const float* hour_emb, const float* day_emb,
    const float* hpad, const float* wpad,
    const int* hmat, const int* wmat, const int* seqlens,
    float* qb, float* kb, float* vb, float* qk, float* qh, float* qw,
    float* aw, float* wsh, float* wsw, float* xattn, float* xln, float* h1)
{
    const int M = BN * LL;
    const long long sBL = (long long)LL * DD;
    // Q, K, V projections
    gemm(st, xin, DD, 0, Wq, DD, 0, 0, bq, nullptr, qb, DD, 0, M, DD, DD, 1, 0);
    gemm(st, xin, DD, 0, Wk, DD, 0, 0, bk, nullptr, kb, DD, 0, M, DD, DD, 1, 0);
    gemm(st, xin, DD, 0, Wv, DD, 0, 0, bv, nullptr, vb, DD, 0, M, DD, DD, 1, 0);
    // scores: Q K^T (batched), plus Q.hour_emb^T / Q.day_emb^T lookup tables
    gemm(st, qb, DD, sBL, kb, DD, sBL, 1, nullptr, nullptr,
         qk, LL, (long long)LL * LL, LL, LL, DD, BN, 0);
    gemm(st, qb, DD, 0, hour_emb, DD, 0, 1, nullptr, nullptr, qh, 32, 0, M, 25, DD, 1, 0);
    gemm(st, qb, DD, 0, day_emb,  DD, 0, 1, nullptr, nullptr, qw, 32, 0, M,  8, DD, 1, 0);
    // softmax with causal + seq-len mask
    k_attn_softmax<<<M, 256, 0, st>>>(qk, qh, qw, hmat, wmat, seqlens, aw);
    // bucket sums for the embedding-weighted aggregation
    k_wsum<<<M, 32, 0, st>>>(aw, hmat, wsh);
    k_wsum<<<M, 32, 0, st>>>(aw, wmat, wsw);
    // x = aw @ V + wsh @ hour_emb + wsw @ day_emb   (K padded to 32)
    gemm(st, aw, LL, (long long)LL * LL, vb, DD, sBL, 0, nullptr, nullptr,
         xattn, DD, sBL, LL, DD, LL, BN, 0);
    gemm(st, wsh, 32, 0, hpad, DD, 0, 0, nullptr, xattn, xattn, DD, 0, M, DD, 32, 1, 0);
    gemm(st, wsw, 32, 0, wpad, DD, 0, 0, nullptr, xattn, xattn, DD, 0, M, DD, 32, 1, 0);
    // residual + LN, FFN, residual + LN
    k_add_ln<<<M, DD, 0, st>>>(xattn, xin, g1, be1, xln);
    gemm(st, xln, DD, 0, W1, DD, 0, 0, b1, nullptr, h1, DD, 0, M, DD, DD, 1, 1);
    gemm(st, h1, DD, 0, W2, DD, 0, 0, b2, nullptr, xattn, DD, 0, M, DD, DD, 1, 0);
    k_add_ln<<<M, DD, 0, st>>>(xattn, xln, g2, be2, xout);
}

// ---------------------------------------------------------------------------
extern "C" void kernel_launch(void* const* d_in, const int* in_sizes, int n_in,
                              void* d_out, int out_size, void* d_ws, size_t ws_size,
                              hipStream_t stream)
{
    (void)in_sizes; (void)n_in; (void)out_size; (void)ws_size;

    // ---- inputs (setup_inputs() dict order, nested dicts in insertion order)
    const float* src      = (const float*)d_in[0];
    const int*   seqlens  = (const int*)  d_in[1];
    const int*   hmat     = (const int*)  d_in[2];
    const int*   wmat     = (const int*)  d_in[3];
    const int*   lhmat    = (const int*)  d_in[4];
    const int*   lwmat    = (const int*)  d_in[5];
    const float* user_emb = (const float*)d_in[6];
    // d_in[7] = batch_seq_idxes (unused by forward)
    const float* hour_emb = (const float*)d_in[8];
    const float* day_emb  = (const float*)d_in[9];
    const float* w1qW = (const float*)d_in[10]; const float* w1qb = (const float*)d_in[11];
    const float* w1kW = (const float*)d_in[12]; const float* w1kb = (const float*)d_in[13];
    const float* w1vW = (const float*)d_in[14]; const float* w1vb = (const float*)d_in[15];
    const float* f1aW = (const float*)d_in[16]; const float* f1ab = (const float*)d_in[17];
    const float* f1bW = (const float*)d_in[18]; const float* f1bb = (const float*)d_in[19];
    const float* ln11g = (const float*)d_in[20]; const float* ln11b = (const float*)d_in[21];
    const float* ln12g = (const float*)d_in[22]; const float* ln12b = (const float*)d_in[23];
    const float* w2qW = (const float*)d_in[24]; const float* w2qb = (const float*)d_in[25];
    const float* w2kW = (const float*)d_in[26]; const float* w2kb = (const float*)d_in[27];
    const float* w2vW = (const float*)d_in[28]; const float* w2vb = (const float*)d_in[29];
    const float* f2aW = (const float*)d_in[30]; const float* f2ab = (const float*)d_in[31];
    const float* f2bW = (const float*)d_in[32]; const float* f2bb = (const float*)d_in[33];
    const float* ln21g = (const float*)d_in[34]; const float* ln21b = (const float*)d_in[35];
    const float* ln22g = (const float*)d_in[36]; const float* ln22b = (const float*)d_in[37];
    const float* upW  = (const float*)d_in[38]; const float* upb  = (const float*)d_in[39];
    const float* decW = (const float*)d_in[40]; const float* decb = (const float*)d_in[41];

    // ---- workspace layout (floats); ~9 MB total
    float* W = (float*)d_ws;
    size_t off = 0;
    auto alloc = [&](size_t n) { float* p = W + off; off += n; return p; };
    const size_t nBLD  = (size_t)BN * LL * DD;
    const size_t nBLL  = (size_t)BN * LL * LL;
    const size_t nBL32 = (size_t)BN * LL * 32;

    float* xpe   = alloc(nBLD);
    float* qb    = alloc(nBLD);
    float* kb    = alloc(nBLD);
    float* vb    = alloc(nBLD);
    float* qk    = alloc(nBLL);
    float* qh    = alloc(nBL32);
    float* qw    = alloc(nBL32);
    float* aw    = alloc(nBLL);
    float* wsh   = alloc(nBL32);
    float* wsw   = alloc(nBL32);
    float* xattn = alloc(nBLD);
    float* xln   = alloc(nBLD);
    float* h1    = alloc(nBLD);
    float* x1    = alloc(nBLD);
    float* x2    = alloc(nBLD);
    float* u     = alloc((size_t)BN * DD);
    float* xu    = alloc((size_t)BN * LL);
    float* hu    = alloc((size_t)BN * 32);
    float* wu    = alloc((size_t)BN * 32);
    float* aw2   = alloc(nBLL);
    float* wsh2  = alloc(nBL32);
    float* wsw2  = alloc(nBL32);
    float* agg   = alloc(nBLD);
    float* hpad  = alloc((size_t)32 * DD);
    float* wpad  = alloc((size_t)32 * DD);

    const int M = BN * LL;
    const long long sBL = (long long)LL * DD;

    // ---- zero-padded [32,128] embedding copies (K=32 GEMMs need no guards)
    k_pad_emb<<<32, DD, 0, stream>>>(hour_emb, 25, hpad);
    k_pad_emb<<<32, DD, 0, stream>>>(day_emb,   8, wpad);

    // ---- positional encoding (batch-index quirk preserved)
    k_add_pe<<<M, DD, 0, stream>>>(src, xpe);

    // ---- two attention blocks
    run_attn_block(stream, xpe, x1,
                   w1qW, w1qb, w1kW, w1kb, w1vW, w1vb,
                   f1aW, f1ab, f1bW, f1bb, ln11g, ln11b, ln12g, ln12b,
                   hour_emb, day_emb, hpad, wpad, hmat, wmat, seqlens,
                   qb, kb, vb, qk, qh, qw, aw, wsh, wsw, xattn, xln, h1);
    run_attn_block(stream, x1, x2,
                   w2qW, w2qb, w2kW, w2kb, w2vW, w2vb,
                   f2aW, f2ab, f2bW, f2bb, ln21g, ln21b, ln22g, ln22b,
                   hour_emb, day_emb, hpad, wpad, hmat, wmat, seqlens,
                   qb, kb, vb, qk, qh, qw, aw, wsh, wsw, xattn, xln, h1);

    // ---- decoder
    // u = user_emb @ u_proj.W + b          [8,128]
    gemm(stream, user_emb, DD, 0, upW, DD, 0, 0, upb, nullptr, u, DD, 0, BN, DD, DD, 1, 0);
    // xu[b,j] = x2[b,j] . u[b]             (batched GEMM, N=1)
    gemm(stream, x2, DD, sBL, u, DD, DD, 1, nullptr, nullptr,
         xu, 1, (long long)LL, LL, 1, DD, BN, 0);
    // hu[b,t] = hour_emb[t] . u[b], wu[b,t] = day_emb[t] . u[b]
    gemm(stream, u, DD, 0, hour_emb, DD, 0, 1, nullptr, nullptr, hu, 32, 0, BN, 25, DD, 1, 0);
    gemm(stream, u, DD, 0, day_emb,  DD, 0, 1, nullptr, nullptr, wu, 32, 0, BN,  8, DD, 1, 0);
    // causal softmax over decoder scores
    k_dec_softmax<<<M, 256, 0, stream>>>(xu, hu, wu, lhmat, lwmat, aw2);
    // bucket sums with label matrices
    k_wsum<<<M, 32, 0, stream>>>(aw2, lhmat, wsh2);
    k_wsum<<<M, 32, 0, stream>>>(aw2, lwmat, wsw2);
    // agg = aw2 @ x2 + wsh2 @ hour_emb + wsw2 @ day_emb   (K padded to 32)
    gemm(stream, aw2, LL, (long long)LL * LL, x2, DD, sBL, 0, nullptr, nullptr,
         agg, DD, sBL, LL, DD, LL, BN, 0);
    gemm(stream, wsh2, 32, 0, hpad, DD, 0, 0, nullptr, agg, agg, DD, 0, M, DD, 32, 1, 0);
    gemm(stream, wsw2, 32, 0, wpad, DD, 0, 0, nullptr, agg, agg, DD, 0, M, DD, 32, 1, 0);
    // final projection: out = agg @ dec.W + dec.b   [1280, 10000]
    gemm(stream, agg, DD, 0, decW, NPOI, 0, 0, decb, nullptr,
         (float*)d_out, NPOI, 0, M, NPOI, DD, 1, 0);
}
